// DVAE_24850680775463
// MI455X (gfx1250) — matrix operand order repaired and checked by reading, not compiled, not run on hardware
//
#include <hip/hip_runtime.h>
#include <cstdint>

// ---------------- problem constants ----------------
constexpr int kB   = 1024;
constexpr int kN   = 16;    // MAXN
constexpr int kNVT = 16;
constexpr int kFS  = 32;
constexpr int kHS  = 512;
constexpr int kNZ  = 64;
constexpr int kVS  = kHS + kN;   // 528
constexpr int kG3  = 3 * kHS;    // 1536

// 16 waves / WG = 4 waves per SIMD32 to hide L2 B-operand latency.
constexpr int kWaves   = 16;
constexpr int kThreads = kWaves * 32;          // 512
constexpr int kSlice   = kHS / kWaves;         // 32 hidden units per wave
constexpr int kTiles   = kSlice / 16;          // 2 N-tiles per gate group

// ---------------- WMMA plumbing (CDNA5 / gfx1250, wave32) ----------------
typedef __attribute__((ext_vector_type(16))) __bf16 bf16x16;
typedef __attribute__((ext_vector_type(8)))  float  f32x8;

union Frag { bf16x16 v; uint4 q[2]; };
static_assert(sizeof(bf16x16) == 32, "bf16x16 must be 32B");

__device__ __forceinline__ f32x8 zero8() {
  f32x8 z = {0.f, 0.f, 0.f, 0.f, 0.f, 0.f, 0.f, 0.f};
  return z;
}

__device__ __forceinline__ f32x8 wmma_bf16(bf16x16 a, bf16x16 b, f32x8 c) {
  // D = A(16x32 bf16) * B(32x16 bf16) + C(16x16 f32)
  return __builtin_amdgcn_wmma_f32_16x16x32_bf16(
      /*neg_a=*/false, a, /*neg_b=*/false, b,
      /*c_mod=*/(short)0, c, /*reuse_a=*/false, /*reuse_b=*/false);
}

// A fragment (16x32, row-major activations, row stride ld elements).
// Layout used here (contiguous K, not the ISA table order): lane<16 holds row (lane&15), K = k0..k0+15 contiguous;
//             lane>=16 holds K = k0+16..k0+31 contiguous.
__device__ __forceinline__ bf16x16 load_a_frag(const __bf16* base, int ld,
                                               int k0, int lane) {
  const int m   = lane & 15;
  const int sel = lane >> 4;
  const __bf16* p = base + m * ld + k0 + sel * 16;
  Frag u;
  u.q[0] = *(const uint4*)(p);        // 8 bf16
  u.q[1] = *(const uint4*)(p + 8);    // 8 bf16, K+8
  return u.v;
}

// B fragment (32x16) built from a row-major weight W[n][k] (so B = W^T).
// Layout used here (contiguous K, same as A): lane holds column (lane&15); lane<16 -> K = k0..k0+15
// contiguous, lane>=16 -> K = k0+16..k0+31 contiguous.
__device__ __forceinline__ bf16x16 load_b_frag(const __bf16* __restrict__ wgt,
                                               int ld, int n0, int k0, int lane) {
  const int n   = lane & 15;
  const int sel = lane >> 4;
  const __bf16* p = wgt + (size_t)(n0 + n) * ld + k0 + sel * 16;
  Frag u;
  u.q[0] = *(const uint4*)(p);
  u.q[1] = *(const uint4*)(p + 8);
  return u.v;
}

// Fast sigmoid/tanh: v_exp_f32 + v_rcp_f32 (avoid IEEE divide expansion).
__device__ __forceinline__ float sigm(float x) {
  return __builtin_amdgcn_rcpf(1.0f + __expf(-x));
}
__device__ __forceinline__ float tanh_fast(float x) {
  return 2.0f * sigm(2.0f * x) - 1.0f;   // tanh(x) = 2*sigmoid(2x) - 1
}

// ---------------- f32 -> bf16 weight conversion ----------------
__global__ void cvt_f32_bf16(const float* __restrict__ src,
                             __bf16* __restrict__ dst, int n) {
  int i = blockIdx.x * blockDim.x + threadIdx.x;
  if (i < n) dst[i] = (__bf16)src[i];
}

// ---------------- main fused kernel ----------------
// One workgroup = 16 batch rows (one WMMA M-tile), 16 waves, each wave owns a
// 32-wide hidden slice (2 N-tiles) for every 512-wide output group.
__global__ __launch_bounds__(kThreads) void dvae_main(
    const int*   __restrict__ types,     // (B,16)
    const float* __restrict__ params,    // (B,16,32)
    const float* __restrict__ adj,       // (B,16,16)
    const float* __restrict__ gt_wi,     // (1536,16)
    const float* __restrict__ gt_bi,     // (1536)
    const float* __restrict__ gt_bh,     // (1536)
    const float* __restrict__ gp_bi,     // (1536)
    const float* __restrict__ gp_bh,     // (1536)
    const float* __restrict__ gate_wf,   // (512,528) f32 (eye column)
    const float* __restrict__ gate_b,    // (512)
    const float* __restrict__ mapper_wf, // (512,528) f32 (eye column)
    const float* __restrict__ fc1_b,     // (64)
    const float* __restrict__ fc2_b,     // (64)
    const __bf16* __restrict__ gtwh,     // (1536,512) bf16
    const __bf16* __restrict__ gpwh,     // (1536,512) bf16
    const __bf16* __restrict__ gpwi,     // (1536,32)  bf16
    const __bf16* __restrict__ gatew,    // (512,528)  bf16
    const __bf16* __restrict__ mapperw,  // (512,528)  bf16
    const __bf16* __restrict__ fc1w,     // (64,512)   bf16
    const __bf16* __restrict__ fc2w,     // (64,512)   bf16
    float* __restrict__ gm,              // (B,16,512) scratch: gate*mapped
    float* __restrict__ out)             // mu (B,64) ++ logvar (B,64)
{
  __shared__ __align__(16) __bf16 hA[16 * kHS];  // current h, bf16 A-tile
  __shared__ __align__(16) float  hF[16 * kHS];  // current h, f32 master
  __shared__ __align__(16) __bf16 pA[16 * kFS];  // params tile, bf16

  const int tid   = threadIdx.x;
  const int w     = tid >> 5;         // wave 0..15
  const int lane  = tid & 31;
  const int b0    = blockIdx.x * 16;  // batch tile
  const int hbase = w * kSlice;       // this wave's hidden slice
  const int ncol  = lane & 15;
  const int sel   = lane >> 4;
  const int mo    = sel * 8;          // row offset of this lane's D elements

  for (int v = 0; v < kN; ++v) {
    // ---- H_v = sum_{n<v} adj[b][v][n] * gm[b][n][:]  (f32 + bf16 copies) --
    for (int i = tid; i < 16 * kHS; i += kThreads) {
      const int m = i >> 9;
      const int h = i & (kHS - 1);
      float acc = 0.f;
      const float* arow = adj + ((size_t)(b0 + m) * kN + v) * kN;
      const float* gmb  = gm + ((size_t)(b0 + m) * kN) * kHS + h;
      for (int n = 0; n < v; ++n) acc += arow[n] * gmb[(size_t)n * kHS];
      hF[i] = acc;
      hA[i] = (__bf16)acc;
    }
    for (int i = tid; i < 16 * kFS; i += kThreads) {
      const int m = i >> 5;
      const int k = i & (kFS - 1);
      pA[i] = (__bf16)params[((size_t)(b0 + m) * kN + v) * kFS + k];
    }
    __syncthreads();

    // ==================== GRU #1: x = one_hot(type) =======================
    {
      f32x8 aR[kTiles], aZ[kTiles], aN[kTiles];
#pragma unroll
      for (int t = 0; t < kTiles; ++t) { aR[t] = zero8(); aZ[t] = zero8(); aN[t] = zero8(); }
#pragma clang loop unroll(disable)
      for (int kc = 0; kc < 16; ++kc) {
        const int k0 = kc * 32;
        const bf16x16 a = load_a_frag(hA, kHS, k0, lane);
#pragma unroll
        for (int t = 0; t < kTiles; ++t) {
          const int n0 = hbase + t * 16;
          aR[t] = wmma_bf16(a, load_b_frag(gtwh, kHS, n0,        k0, lane), aR[t]);
          aZ[t] = wmma_bf16(a, load_b_frag(gtwh, kHS, n0 + 512,  k0, lane), aZ[t]);
          aN[t] = wmma_bf16(a, load_b_frag(gtwh, kHS, n0 + 1024, k0, lane), aN[t]);
        }
      }
      __syncthreads();  // all reads of hA complete
#pragma unroll
      for (int t = 0; t < kTiles; ++t) {
        const int hcol = hbase + t * 16 + ncol;
#pragma unroll
        for (int j = 0; j < 8; ++j) {
          const int m  = mo + j;
          const int ty = types[(b0 + m) * kN + v];
          // gi = one_hot @ gt_wi.T + gt_bi  == column gather
          const float gr = gt_wi[(size_t)(hcol       ) * kNVT + ty] + gt_bi[hcol       ];
          const float gz = gt_wi[(size_t)(hcol +  512) * kNVT + ty] + gt_bi[hcol +  512];
          const float gn = gt_wi[(size_t)(hcol + 1024) * kNVT + ty] + gt_bi[hcol + 1024];
          const float hr = aR[t][j] + gt_bh[hcol       ];
          const float hz = aZ[t][j] + gt_bh[hcol +  512];
          const float hn = aN[t][j] + gt_bh[hcol + 1024];
          const float r  = sigm(gr + hr);
          const float z  = sigm(gz + hz);
          const float nn = tanh_fast(gn + r * hn);
          const float hp = hF[m * kHS + hcol];
          const float hv = (1.f - z) * nn + z * hp;
          hF[m * kHS + hcol] = hv;
          hA[m * kHS + hcol] = (__bf16)hv;
        }
      }
      __syncthreads();
    }

    // ==================== GRU #2: x = params ==============================
    {
      f32x8 aR[kTiles], aZ[kTiles], aN[kTiles];
#pragma unroll
      for (int t = 0; t < kTiles; ++t) { aR[t] = zero8(); aZ[t] = zero8(); aN[t] = zero8(); }
#pragma clang loop unroll(disable)
      for (int kc = 0; kc < 16; ++kc) {
        const int k0 = kc * 32;
        const bf16x16 a = load_a_frag(hA, kHS, k0, lane);
#pragma unroll
        for (int t = 0; t < kTiles; ++t) {
          const int n0 = hbase + t * 16;
          aR[t] = wmma_bf16(a, load_b_frag(gpwh, kHS, n0,        k0, lane), aR[t]);
          aZ[t] = wmma_bf16(a, load_b_frag(gpwh, kHS, n0 + 512,  k0, lane), aZ[t]);
          aN[t] = wmma_bf16(a, load_b_frag(gpwh, kHS, n0 + 1024, k0, lane), aN[t]);
        }
      }
      __syncthreads();
      // gi = params(16x32) @ gp_wi.T : one K=32 WMMA per N-tile (reads pA only)
      const bf16x16 ap = load_a_frag(pA, kFS, 0, lane);
      f32x8 gR[kTiles], gZ[kTiles], gN[kTiles];
#pragma unroll
      for (int t = 0; t < kTiles; ++t) {
        const int n0 = hbase + t * 16;
        gR[t] = wmma_bf16(ap, load_b_frag(gpwi, kFS, n0,        0, lane), zero8());
        gZ[t] = wmma_bf16(ap, load_b_frag(gpwi, kFS, n0 + 512,  0, lane), zero8());
        gN[t] = wmma_bf16(ap, load_b_frag(gpwi, kFS, n0 + 1024, 0, lane), zero8());
      }
#pragma unroll
      for (int t = 0; t < kTiles; ++t) {
        const int hcol = hbase + t * 16 + ncol;
#pragma unroll
        for (int j = 0; j < 8; ++j) {
          const int m = mo + j;
          const float gr = gR[t][j] + gp_bi[hcol       ];
          const float gz = gZ[t][j] + gp_bi[hcol +  512];
          const float gn = gN[t][j] + gp_bi[hcol + 1024];
          const float hr = aR[t][j] + gp_bh[hcol       ];
          const float hz = aZ[t][j] + gp_bh[hcol +  512];
          const float hn = aN[t][j] + gp_bh[hcol + 1024];
          const float r  = sigm(gr + hr);
          const float z  = sigm(gz + hz);
          const float nn = tanh_fast(gn + r * hn);
          const float hp = hF[m * kHS + hcol];
          const float hv = (1.f - z) * nn + z * hp;
          hF[m * kHS + hcol] = hv;
          hA[m * kHS + hcol] = (__bf16)hv;
        }
      }
      __syncthreads();
    }

    // ============ gate/mapper for node v  ->  gm[b][v][:] =================
    // G_aug = [hv, e_v]: K=512 WMMA chain + scalar add of column (512+v).
    {
      f32x8 ag[kTiles], am[kTiles];
#pragma unroll
      for (int t = 0; t < kTiles; ++t) { ag[t] = zero8(); am[t] = zero8(); }
#pragma clang loop unroll(disable)
      for (int kc = 0; kc < 16; ++kc) {
        const int k0 = kc * 32;
        const bf16x16 a = load_a_frag(hA, kHS, k0, lane);
#pragma unroll
        for (int t = 0; t < kTiles; ++t) {
          const int n0 = hbase + t * 16;
          ag[t] = wmma_bf16(a, load_b_frag(gatew,   kVS, n0, k0, lane), ag[t]);
          am[t] = wmma_bf16(a, load_b_frag(mapperw, kVS, n0, k0, lane), am[t]);
        }
      }
#pragma unroll
      for (int t = 0; t < kTiles; ++t) {
        const int hcol = hbase + t * 16 + ncol;
        const float ge = gate_wf  [(size_t)hcol * kVS + 512 + v] + gate_b[hcol];
        const float me = mapper_wf[(size_t)hcol * kVS + 512 + v];
#pragma unroll
        for (int j = 0; j < 8; ++j) {
          const int m = mo + j;
          const float g  = sigm(ag[t][j] + ge);
          const float mp = am[t][j] + me;
          gm[((size_t)(b0 + m) * kN + v) * kHS + hcol] = g * mp;
        }
      }
      __threadfence_block();  // gm producer/consumer share this WGP
      __syncthreads();
    }
  }

  // =================== FC head: mu / logvar from hv15 =====================
  if (w < 8) {
    const int mat = w >> 2;            // 0 = mu (fc1), 1 = logvar (fc2)
    const int t   = w & 3;             // N-tile 0..3 of 64 outputs
    const __bf16* fw = mat ? fc2w : fc1w;
    const float*  fb = mat ? fc2_b : fc1_b;
    f32x8 acc = zero8();
#pragma clang loop unroll(disable)
    for (int kc = 0; kc < 16; ++kc) {
      const int k0 = kc * 32;
      acc = wmma_bf16(load_a_frag(hA, kHS, k0, lane),
                      load_b_frag(fw, kHS, t * 16, k0, lane), acc);
    }
    const int col = t * 16 + ncol;
    const float bb = fb[col];
#pragma unroll
    for (int j = 0; j < 8; ++j) {
      const int m = mo + j;
      out[(size_t)mat * kB * kNZ + (size_t)(b0 + m) * kNZ + col] = acc[j] + bb;
    }
  }
}

// ---------------- launch ----------------
extern "C" void kernel_launch(void* const* d_in, const int* in_sizes, int n_in,
                              void* d_out, int out_size, void* d_ws, size_t ws_size,
                              hipStream_t stream) {
  const int*   types    = (const int*)  d_in[0];
  const float* params   = (const float*)d_in[1];
  const float* adj      = (const float*)d_in[2];
  const float* gt_wi    = (const float*)d_in[3];
  const float* gt_wh    = (const float*)d_in[4];
  const float* gt_bi    = (const float*)d_in[5];
  const float* gt_bh    = (const float*)d_in[6];
  const float* gp_wi    = (const float*)d_in[7];
  const float* gp_wh    = (const float*)d_in[8];
  const float* gp_bi    = (const float*)d_in[9];
  const float* gp_bh    = (const float*)d_in[10];
  const float* gate_w   = (const float*)d_in[11];
  const float* gate_b   = (const float*)d_in[12];
  const float* mapper_w = (const float*)d_in[13];
  const float* fc1_w    = (const float*)d_in[14];
  const float* fc1_b    = (const float*)d_in[15];
  const float* fc2_w    = (const float*)d_in[16];
  const float* fc2_b    = (const float*)d_in[17];

  // workspace layout: [gm f32 (B,16,512)] [bf16 weight copies]
  char*  ws  = (char*)d_ws;
  float* gm  = (float*)ws;
  size_t off = (size_t)kB * kN * kHS * sizeof(float);
  __bf16* gtwh_b = (__bf16*)(ws + off); off += (size_t)kG3 * kHS * 2;
  __bf16* gpwh_b = (__bf16*)(ws + off); off += (size_t)kG3 * kHS * 2;
  __bf16* gpwi_b = (__bf16*)(ws + off); off += (size_t)kG3 * kFS * 2;
  __bf16* gate_bb = (__bf16*)(ws + off); off += (size_t)kHS * kVS * 2;
  __bf16* mapp_bb = (__bf16*)(ws + off); off += (size_t)kHS * kVS * 2;
  __bf16* fc1_bb  = (__bf16*)(ws + off); off += (size_t)kNZ * kHS * 2;
  __bf16* fc2_bb  = (__bf16*)(ws + off); off += (size_t)kNZ * kHS * 2;

  auto cvt = [&](const float* s, __bf16* d, int n) {
    cvt_f32_bf16<<<(n + 255) / 256, 256, 0, stream>>>(s, d, n);
  };
  cvt(gt_wh,    gtwh_b,  kG3 * kHS);
  cvt(gp_wh,    gpwh_b,  kG3 * kHS);
  cvt(gp_wi,    gpwi_b,  kG3 * kFS);
  cvt(gate_w,   gate_bb, kHS * kVS);
  cvt(mapper_w, mapp_bb, kHS * kVS);
  cvt(fc1_w,    fc1_bb,  kNZ * kHS);
  cvt(fc2_w,    fc2_bb,  kNZ * kHS);

  dvae_main<<<kB / 16, kThreads, 0, stream>>>(
      types, params, adj, gt_wi, gt_bi, gt_bh, gp_bi, gp_bh,
      gate_w, gate_b, mapper_w, fc1_b, fc2_b,
      gtwh_b, gpwh_b, gpwi_b, gate_bb, mapp_bb, fc1_bb, fc2_bb,
      gm, (float*)d_out);
}
